// SimpleAttentionWithKVCache_23029614641155
// MI455X (gfx1250) — compile-verified, matrix operand order checked by reading, not run on hardware
//
#include <hip/hip_runtime.h>

// MI455X / gfx1250, wave32. All matmuls via v_wmma_f32_16x16x32_bf16.

typedef __attribute__((ext_vector_type(16))) __bf16 v16bf;
typedef __attribute__((ext_vector_type(8)))  float  v8f;

#define BB 2
#define SS 2048
#define DD 1024
#define HH 16
#define HD 64

// ---------------------------------------------------------------------------
// Fragment loader: 16-bit WMMA A/B fragments are symmetric for K-contiguous
// data. Caller passes p = base + row*stride + koff where row includes
// (lane&15) and koff includes (lane>>4)*8. VGPRs 0-3 <- K[0..7] (16B at p),
// VGPRs 4-7 <- K[16..23] (16B at p+16 elements).
// ---------------------------------------------------------------------------
__device__ __forceinline__ v16bf load_frag(const __bf16* p) {
    union { v16bf v; uint4 u[2]; } t;
    t.u[0] = *(const uint4*)(p);
    t.u[1] = *(const uint4*)(p + 16);
    return t.v;
}

__device__ __forceinline__ v8f wmma_bf16(v16bf a, v16bf b, v8f c) {
    return __builtin_amdgcn_wmma_f32_16x16x32_bf16(
        false, a, false, b, (short)0, c, false, false);
}

// ---------------------------------------------------------------------------
// fp32 -> bf16 conversion, 4 elements/thread
// ---------------------------------------------------------------------------
struct bf16x4 { __bf16 a, b, c, d; };

__global__ __launch_bounds__(256) void f32_to_bf16_kernel(
    const float* __restrict__ in, __bf16* __restrict__ out, int n4)
{
    int i = blockIdx.x * blockDim.x + threadIdx.x;
    if (i < n4) {
        float4 f = ((const float4*)in)[i];
        bf16x4 o = { (__bf16)f.x, (__bf16)f.y, (__bf16)f.z, (__bf16)f.w };
        ((bf16x4*)out)[i] = o;
    }
}

// ---------------------------------------------------------------------------
// GEMM: C[M,N] = A[M,K] @ W[N,K]^T + bias[N]
// 64x64 tile per wave, 4 waves/block. K-loop step 32.
// mode 0: Q  -> bf16 [B,H,S,Hd], scaled by 1/sqrt(Hd)
// mode 1: K  -> bf16 [B,H,S,Hd]
// mode 2: V  -> bf16 [B,H,Hd,S]  (transposed for PV fragment loads)
// mode 3: f32 row-major [M,N] (final output)
// ---------------------------------------------------------------------------
__global__ __launch_bounds__(128) void gemm_bf16_kernel(
    const __bf16* __restrict__ A, const __bf16* __restrict__ W,
    const float* __restrict__ bias, void* __restrict__ out,
    int M, int N, int K, int mode, float scale)
{
    const int wave = threadIdx.x >> 5;
    const int lane = threadIdx.x & 31;
    const int wid  = blockIdx.x * 4 + wave;
    const int tiles_n = N >> 6;
    const int m0 = (wid / tiles_n) << 6;
    const int n0 = (wid % tiles_n) << 6;
    const int lrow  = lane & 15;
    const int lko   = (lane >> 4) * 8;
    const int rbase = (lane >> 4) * 8;

    v8f acc[4][4];
#pragma unroll
    for (int mt = 0; mt < 4; ++mt)
#pragma unroll
        for (int nt = 0; nt < 4; ++nt) acc[mt][nt] = (v8f){};

    for (int k = 0; k < K; k += 32) {
        v16bf af[4], bf[4];
#pragma unroll
        for (int mt = 0; mt < 4; ++mt)
            af[mt] = load_frag(A + (size_t)(m0 + mt * 16 + lrow) * K + k + lko);
#pragma unroll
        for (int nt = 0; nt < 4; ++nt)
            bf[nt] = load_frag(W + (size_t)(n0 + nt * 16 + lrow) * K + k + lko);
#pragma unroll
        for (int mt = 0; mt < 4; ++mt)
#pragma unroll
            for (int nt = 0; nt < 4; ++nt)
                acc[mt][nt] = wmma_bf16(af[mt], bf[nt], acc[mt][nt]);
    }

#pragma unroll
    for (int mt = 0; mt < 4; ++mt) {
#pragma unroll
        for (int nt = 0; nt < 4; ++nt) {
#pragma unroll
            for (int r = 0; r < 8; ++r) {
                const int m = m0 + mt * 16 + rbase + r;
                const int n = n0 + nt * 16 + lrow;
                float val = acc[mt][nt][r] + bias[n];
                if (mode == 3) {
                    ((float*)out)[(size_t)m * N + n] = val;
                } else {
                    const int b = m / SS, s = m % SS;
                    const int h = n / HD, d = n % HD;
                    __bf16* o = (__bf16*)out;
                    if (mode == 0)
                        o[(((size_t)b * HH + h) * SS + s) * HD + d] = (__bf16)(val * scale);
                    else if (mode == 1)
                        o[(((size_t)b * HH + h) * SS + s) * HD + d] = (__bf16)val;
                    else
                        o[(((size_t)b * HH + h) * HD + d) * SS + s] = (__bf16)val;
                }
            }
        }
    }
}

// ---------------------------------------------------------------------------
// Flash attention (causal). Q pre-scaled by 1/sqrt(Hd).
// One wave = 32 query rows x full head. Key tiles of 32, online softmax.
// Q/K: [B*H, S, Hd] bf16.  Vt: [B*H, Hd, S] bf16.  Out: [B, S, D] bf16.
// P (C-layout f32) is re-striped to A-layout via per-wave private LDS.
// ---------------------------------------------------------------------------
__global__ __launch_bounds__(128) void flash_attn_kernel(
    const __bf16* __restrict__ Q, const __bf16* __restrict__ Km,
    const __bf16* __restrict__ Vt, __bf16* __restrict__ O)
{
    __shared__ __bf16 plds_all[4][32 * 32];   // 2 KB per wave, private

    const int wave = threadIdx.x >> 5;
    const int lane = threadIdx.x & 31;
    const int wid  = blockIdx.x * 4 + wave;
    const int QT   = SS / 32;                 // 64 query tiles per head
    const int bh   = wid / QT;
    const int qt   = wid % QT;
    const int q0   = qt * 32;
    const int lrow  = lane & 15;
    const int lhalf = lane >> 4;
    const int lko   = lhalf * 8;

    const __bf16* qp = Q  + (size_t)bh * SS * HD;
    const __bf16* kp = Km + (size_t)bh * SS * HD;
    const __bf16* vp = Vt + (size_t)bh * HD * SS;
    __bf16* pl = plds_all[wave];

    // resident Q fragments: 2 M-subtiles x 2 K-steps over Hd=64
    v16bf qf[2][2];
#pragma unroll
    for (int mt = 0; mt < 2; ++mt)
#pragma unroll
        for (int ks = 0; ks < 2; ++ks)
            qf[mt][ks] = load_frag(qp + (size_t)(q0 + mt * 16 + lrow) * HD + ks * 32 + lko);

    v8f o[2][4];
#pragma unroll
    for (int mt = 0; mt < 2; ++mt)
#pragma unroll
        for (int dt = 0; dt < 4; ++dt) o[mt][dt] = (v8f){};

    float mrun[2][8], lrun[2][8];
#pragma unroll
    for (int mt = 0; mt < 2; ++mt)
#pragma unroll
        for (int r = 0; r < 8; ++r) { mrun[mt][r] = -1e30f; lrun[mt][r] = 0.0f; }

    const int nkt = qt + 1;                   // causal: key tiles 0..qt
    for (int kt = 0; kt < nkt; ++kt) {
        const int k0 = kt * 32;

        // ---- scores S = Q @ K^T (32x32 tile) ----
        v8f s[2][2];
#pragma unroll
        for (int mt = 0; mt < 2; ++mt)
#pragma unroll
            for (int nt = 0; nt < 2; ++nt) s[mt][nt] = (v8f){};
#pragma unroll
        for (int ks = 0; ks < 2; ++ks) {
            v16bf kf0 = load_frag(kp + (size_t)(k0 + lrow) * HD + ks * 32 + lko);
            v16bf kf1 = load_frag(kp + (size_t)(k0 + 16 + lrow) * HD + ks * 32 + lko);
#pragma unroll
            for (int mt = 0; mt < 2; ++mt) {
                s[mt][0] = wmma_bf16(qf[mt][ks], kf0, s[mt][0]);
                s[mt][1] = wmma_bf16(qf[mt][ks], kf1, s[mt][1]);
            }
        }

        // ---- causal mask on the diagonal tile only ----
        if (kt == qt) {
#pragma unroll
            for (int mt = 0; mt < 2; ++mt)
#pragma unroll
                for (int nt = 0; nt < 2; ++nt)
#pragma unroll
                    for (int r = 0; r < 8; ++r) {
                        const int qi = q0 + mt * 16 + lhalf * 8 + r;
                        const int ki = k0 + nt * 16 + lrow;
                        if (ki > qi) s[mt][nt][r] = -1e30f;
                    }
        }

        // ---- online softmax update (row stats via wave32 shfl_xor) ----
#pragma unroll
        for (int mt = 0; mt < 2; ++mt) {
#pragma unroll
            for (int r = 0; r < 8; ++r) {
                float tmax = fmaxf(s[mt][0][r], s[mt][1][r]);
                tmax = fmaxf(tmax, __shfl_xor(tmax, 1, 32));
                tmax = fmaxf(tmax, __shfl_xor(tmax, 2, 32));
                tmax = fmaxf(tmax, __shfl_xor(tmax, 4, 32));
                tmax = fmaxf(tmax, __shfl_xor(tmax, 8, 32));
                const float mnew  = fmaxf(mrun[mt][r], tmax);
                const float alpha = __expf(mrun[mt][r] - mnew);
                mrun[mt][r] = mnew;
#pragma unroll
                for (int dt = 0; dt < 4; ++dt) o[mt][dt][r] *= alpha;
                const float p0 = __expf(s[mt][0][r] - mnew);
                const float p1 = __expf(s[mt][1][r] - mnew);
                s[mt][0][r] = p0; s[mt][1][r] = p1;
                float sum = p0 + p1;
                sum += __shfl_xor(sum, 1, 32);
                sum += __shfl_xor(sum, 2, 32);
                sum += __shfl_xor(sum, 4, 32);
                sum += __shfl_xor(sum, 8, 32);
                lrun[mt][r] = lrun[mt][r] * alpha + sum;
            }
        }

        // ---- re-stripe P: C-layout -> A-layout via private LDS ----
#pragma unroll
        for (int mt = 0; mt < 2; ++mt)
#pragma unroll
            for (int nt = 0; nt < 2; ++nt)
#pragma unroll
                for (int r = 0; r < 8; ++r)
                    pl[(mt * 16 + lhalf * 8 + r) * 32 + nt * 16 + lrow] =
                        (__bf16)s[mt][nt][r];
        // same-wave DS ordering: compiler inserts s_wait_dscnt before reads

        // ---- O += P @ V (K=32, one WMMA step; V is pre-transposed) ----
#pragma unroll
        for (int mt = 0; mt < 2; ++mt) {
            v16bf pf = load_frag(pl + (mt * 16 + lrow) * 32 + lko);
#pragma unroll
            for (int dt = 0; dt < 4; ++dt) {
                v16bf vf = load_frag(vp + (size_t)(dt * 16 + lrow) * SS + k0 + lko);
                o[mt][dt] = wmma_bf16(pf, vf, o[mt][dt]);
            }
        }
    }

    // ---- normalize and store attn as bf16 [B, S, D] ----
    const int b = bh / HH, h = bh % HH;
#pragma unroll
    for (int mt = 0; mt < 2; ++mt) {
#pragma unroll
        for (int r = 0; r < 8; ++r) {
            const float inv = 1.0f / lrun[mt][r];
            const int srow = q0 + mt * 16 + lhalf * 8 + r;
#pragma unroll
            for (int dt = 0; dt < 4; ++dt) {
                const int col = h * HD + dt * 16 + lrow;
                O[((size_t)b * SS + srow) * DD + col] = (__bf16)(o[mt][dt][r] * inv);
            }
        }
    }
}

// ---------------------------------------------------------------------------
extern "C" void kernel_launch(void* const* d_in, const int* in_sizes, int n_in,
                              void* d_out, int out_size, void* d_ws, size_t ws_size,
                              hipStream_t stream) {
    const float* x  = (const float*)d_in[0];
    const float* wq = (const float*)d_in[1];
    const float* bq = (const float*)d_in[2];
    const float* wk = (const float*)d_in[3];
    const float* bk = (const float*)d_in[4];
    const float* wv = (const float*)d_in[5];
    const float* bv = (const float*)d_in[6];
    const float* wo = (const float*)d_in[7];
    const float* bo = (const float*)d_in[8];

    const size_t M  = (size_t)BB * SS;      // 4096
    const size_t DW = (size_t)DD * DD;      // 1M
    const size_t QS = (size_t)BB * HH * SS * HD; // 4M

    __bf16* wsb   = (__bf16*)d_ws;
    size_t off = 0;
    __bf16* xb    = wsb + off; off += M * DD;
    __bf16* wqb   = wsb + off; off += DW;
    __bf16* wkb   = wsb + off; off += DW;
    __bf16* wvb   = wsb + off; off += DW;
    __bf16* wob   = wsb + off; off += DW;
    __bf16* qb    = wsb + off; off += QS;
    __bf16* kb    = wsb + off; off += QS;
    __bf16* vtb   = wsb + off; off += QS;
    __bf16* attnb = wsb + off; off += M * DD;

    // 1) fp32 -> bf16 staging
    {
        int n4 = (int)(M * DD / 4);
        f32_to_bf16_kernel<<<(n4 + 255) / 256, 256, 0, stream>>>(x, xb, n4);
        int w4 = (int)(DW / 4);
        f32_to_bf16_kernel<<<(w4 + 255) / 256, 256, 0, stream>>>(wq, wqb, w4);
        f32_to_bf16_kernel<<<(w4 + 255) / 256, 256, 0, stream>>>(wk, wkb, w4);
        f32_to_bf16_kernel<<<(w4 + 255) / 256, 256, 0, stream>>>(wv, wvb, w4);
        f32_to_bf16_kernel<<<(w4 + 255) / 256, 256, 0, stream>>>(wo, wob, w4);
    }

    // 2) Q/K/V projections: (M/64)*(D/64) = 1024 wave-tiles / 4 waves = 256 blocks
    const float qscale = 0.125f;            // 1/sqrt(64)
    gemm_bf16_kernel<<<256, 128, 0, stream>>>(xb, wqb, bq, qb,  (int)M, DD, DD, 0, qscale);
    gemm_bf16_kernel<<<256, 128, 0, stream>>>(xb, wkb, bk, kb,  (int)M, DD, DD, 1, 1.0f);
    gemm_bf16_kernel<<<256, 128, 0, stream>>>(xb, wvb, bv, vtb, (int)M, DD, DD, 2, 1.0f);

    // 3) causal flash attention: 32 heads * 64 qtiles = 2048 waves / 4 = 512 blocks
    flash_attn_kernel<<<512, 128, 0, stream>>>(qb, kb, vtb, attnb);

    // 4) output projection -> fp32 d_out
    gemm_bf16_kernel<<<256, 128, 0, stream>>>(attnb, wob, bo, d_out, (int)M, DD, DD, 3, 1.0f);
}
